// CachedAttentionBlock_60095182405851
// MI455X (gfx1250) — matrix operand order repaired and checked by reading, not compiled, not run on hardware
//
#include <hip/hip_runtime.h>

// ---------------------------------------------------------------------------
// Problem constants (from reference)
// ---------------------------------------------------------------------------
static constexpr int Bb = 4;
static constexpr int Ss = 2048;
static constexpr int Dm = 1024;   // d_model
static constexpr int Hh = 16;     // heads
static constexpr int TOK = Bb * Ss;         // 8192 tokens
static constexpr float EPSV = 1e-6f;

// ---------------------------------------------------------------------------
// WMMA / TDM types (gfx1250, wave32)
// ---------------------------------------------------------------------------
typedef __attribute__((ext_vector_type(16))) __bf16 v16bf;
typedef __attribute__((ext_vector_type(8)))  float  v8f;
typedef __attribute__((ext_vector_type(4)))  unsigned int u32x4;
typedef __attribute__((ext_vector_type(8)))  int i32x8;
typedef __attribute__((ext_vector_type(4)))  int i32x4;

union Frag16 {            // 16 bf16 lanes-worth = 32 bytes = 8 VGPRs
    v16bf v;
    uint4 q[2];
};

static __device__ __forceinline__ unsigned short f2bf(float f) {
    unsigned int u = __float_as_uint(f);
    u += 0x7FFFu + ((u >> 16) & 1u);      // round-to-nearest-even
    return (unsigned short)(u >> 16);
}

static __device__ __forceinline__ v8f wmma_bf16(const Frag16& a, const Frag16& b, v8f c) {
    return __builtin_amdgcn_wmma_f32_16x16x32_bf16(
        false, a.v, false, b.v, (short)0, c, false, false);
}

static __device__ __forceinline__ unsigned lds_off(const void* p) {
    // LDS aperture: low 32 bits of the generic address are the LDS byte offset
    return (unsigned)(unsigned long long)p;
}

#if __has_builtin(__builtin_amdgcn_tensor_load_to_lds)
#define HAVE_TDM 1
#else
#define HAVE_TDM 0
#endif

#if HAVE_TDM
// 2D tile load via Tensor Data Mover: bf16 elements, row stride in elements.
static __device__ __forceinline__ void tdm_load_2d_bf16(
    unsigned ldsByteOff, const void* gptr,
    unsigned tensorD0, unsigned tensorD1,
    unsigned long long strideElems,
    unsigned tileD0, unsigned tileD1) {
    const unsigned long long ga = (unsigned long long)gptr;
    u32x4 g0;
    g0[0] = 1u;                                           // count=1 (valid user D#)
    g0[1] = ldsByteOff;                                   // lds_addr
    g0[2] = (unsigned)(ga & 0xFFFFFFFFu);                 // global_addr[31:0]
    g0[3] = (unsigned)((ga >> 32) & 0x01FFFFFFu) | (2u << 30);  // addr[56:32] | type=2
    i32x8 g1;
    g1[0] = (int)(1u << 16);                              // wg_mask=0, data_size=1 (2B)
    g1[1] = (int)((tensorD0 & 0xFFFFu) << 16);            // tensor_dim0 lo16
    g1[2] = (int)(((tensorD0 >> 16) & 0xFFFFu) | ((tensorD1 & 0xFFFFu) << 16));
    g1[3] = (int)(((tensorD1 >> 16) & 0xFFFFu) | ((tileD0 & 0xFFFFu) << 16));
    g1[4] = (int)(tileD1 & 0xFFFFu);                      // tile_dim1, tile_dim2=0
    g1[5] = (int)(strideElems & 0xFFFFFFFFull);           // dim0_stride lo32
    g1[6] = (int)((strideElems >> 32) & 0xFFFFull);       // dim0_stride hi16
    g1[7] = 0;
    i32x4 gz = {0, 0, 0, 0};
#if defined(__clang_major__) && (__clang_major__ >= 23)
    i32x8 gz8 = {0, 0, 0, 0, 0, 0, 0, 0};
    __builtin_amdgcn_tensor_load_to_lds(g0, g1, gz, gz, gz8, 0);
#else
    __builtin_amdgcn_tensor_load_to_lds(g0, g1, gz, gz, 0);
#endif
}
#endif

static __device__ __forceinline__ void async_load_b128_to_lds(
    unsigned ldsByteOff, const void* gptr) {
    const unsigned long long ga = (unsigned long long)gptr;
    asm volatile("global_load_async_to_lds_b128 %0, %1, off"
                 :: "v"(ldsByteOff), "v"(ga) : "memory");
}

// ---------------------------------------------------------------------------
// Kernel 1: fp32 [R x C] -> bf16 transposed [C x R]  (weight prep)
// ---------------------------------------------------------------------------
__global__ __launch_bounds__(256)
void transpose_bf16_kernel(const float* __restrict__ src,
                           unsigned short* __restrict__ dst, int R, int C) {
    int idx = blockIdx.x * 256 + threadIdx.x;
    if (idx >= R * C) return;
    int r = idx / C, c = idx - r * C;
    dst[(size_t)c * R + r] = f2bf(src[idx]);
}

// ---------------------------------------------------------------------------
// Kernel 2: per-token LayerNorm (+ optional RoPE over full d_model) -> bf16
// ---------------------------------------------------------------------------
__global__ __launch_bounds__(256)
void ln_rope_kernel(const float* __restrict__ x,
                    unsigned short* __restrict__ out, int doRope) {
    __shared__ float xs[Dm];
    __shared__ float wsum[8], wsum2[8];
    const int tid  = threadIdx.x;
    const int lane = tid & 31, wave = tid >> 5;
    const size_t base = (size_t)blockIdx.x * Dm;

    float4 v = *(const float4*)(x + base + tid * 4);
    xs[tid * 4 + 0] = v.x; xs[tid * 4 + 1] = v.y;
    xs[tid * 4 + 2] = v.z; xs[tid * 4 + 3] = v.w;
    float s  = v.x + v.y + v.z + v.w;
    float s2 = v.x * v.x + v.y * v.y + v.z * v.z + v.w * v.w;
    #pragma unroll
    for (int off = 16; off; off >>= 1) {
        s  += __shfl_xor(s,  off, 32);
        s2 += __shfl_xor(s2, off, 32);
    }
    if (lane == 0) { wsum[wave] = s; wsum2[wave] = s2; }
    __syncthreads();
    float mu = 0.f, ms = 0.f;
    #pragma unroll
    for (int w = 0; w < 8; ++w) { mu += wsum[w]; ms += wsum2[w]; }
    mu *= (1.f / Dm); ms *= (1.f / Dm);
    const float rstd = rsqrtf(ms - mu * mu + EPSV);

    #pragma unroll
    for (int i = 0; i < 4; ++i) {
        int d = tid * 4 + i;
        xs[d] = (xs[d] - mu) * rstd;
    }
    __syncthreads();

    if (doRope) {
        if (tid < 128) {
            const int spos = blockIdx.x & (Ss - 1);
            #pragma unroll
            for (int i = 0; i < 4; ++i) {
                int d = tid * 4 + i;          // d in [0,512)
                float invf = __powf(10000.f, -(float)d * (1.f / 512.f));
                float ang  = (float)spos * invf;
                float sn = __sinf(ang), cs = __cosf(ang);
                float x1 = xs[d], x2 = xs[d + 512];
                out[base + d]       = f2bf(x1 * cs - x2 * sn);
                out[base + d + 512] = f2bf(x2 * cs + x1 * sn);
            }
        }
    } else {
        #pragma unroll
        for (int i = 0; i < 4; ++i) {
            int d = tid * 4 + i;
            out[base + d] = f2bf(xs[d]);
        }
    }
}

// ---------------------------------------------------------------------------
// Kernel 3/5: WMMA GEMM  C[M,N] = (A[M,K] * BT[N,K]^T + bias[N]) * outScale
// 256 threads = 8 waves in 4(M) x 2(N); wave tile 16x32; block tile 64x64
// ---------------------------------------------------------------------------
template <bool F32OUT>
__global__ __launch_bounds__(256)
void wmma_gemm_kernel(const unsigned short* __restrict__ A,
                      const unsigned short* __restrict__ BT,
                      const float* __restrict__ bias,
                      void* __restrict__ Cout, int M, int N, int K,
                      float outScale) {
    const int tid  = threadIdx.x;
    const int lane = tid & 31;
    const int wave = tid >> 5;
    const int ln = lane & 15, hs = lane >> 4;
    const int wm = wave >> 1, wn = wave & 1;
    const int m0 = blockIdx.x * 64 + wm * 16;
    const int n0 = blockIdx.y * 64 + wn * 32;

    const unsigned short* pa  = A  + (size_t)(m0 + ln) * K;
    const unsigned short* pb0 = BT + (size_t)(n0 + ln) * K;
    const unsigned short* pb1 = BT + (size_t)(n0 + 16 + ln) * K;

    v8f acc0 = {}, acc1 = {};
    for (int k0 = 0; k0 < K; k0 += 32) {
        if (k0 + 128 < K) {
            __builtin_prefetch(pa  + k0 + 128, 0, 1);
            __builtin_prefetch(pb0 + k0 + 128, 0, 1);
            __builtin_prefetch(pb1 + k0 + 128, 0, 1);
        }
        Frag16 a, b0, b1;
        a.q[0]  = *(const uint4*)(pa  + k0 + hs * 16);
        a.q[1]  = *(const uint4*)(pa  + k0 + hs * 16 + 8);
        b0.q[0] = *(const uint4*)(pb0 + k0 + hs * 16);
        b0.q[1] = *(const uint4*)(pb0 + k0 + hs * 16 + 8);
        b1.q[0] = *(const uint4*)(pb1 + k0 + hs * 16);
        b1.q[1] = *(const uint4*)(pb1 + k0 + hs * 16 + 8);
        acc0 = wmma_bf16(a, b0, acc0);
        acc1 = wmma_bf16(a, b1, acc1);
    }

    #pragma unroll
    for (int t = 0; t < 2; ++t) {
        const int col = n0 + t * 16 + ln;
        const float bvl = bias ? bias[col] : 0.f;
        const v8f& acc = t ? acc1 : acc0;
        #pragma unroll
        for (int r = 0; r < 8; ++r) {
            const int row = m0 + hs * 8 + r;
            const float val = (acc[r] + bvl) * outScale;
            if (F32OUT)
                ((float*)Cout)[(size_t)row * N + col] = val;
            else
                ((unsigned short*)Cout)[(size_t)row * N + col] = f2bf(val);
        }
    }
}

// ---------------------------------------------------------------------------
// Kernel 4: fused flash attention, computed TRANSPOSED (S^T = K Q^T,
// ctx^T = V^T P^T) so each lane owns one q-row: softmax stats are per-lane
// scalars, max/sum need a single shfl_xor(16), and P^T needs only a 4-dword
// half-wave swap to become a WMMA B-fragment (no LDS round trip).
// K tile staged by TDM, V tile by async-to-LDS; both double-buffered and
// issued one tile ahead (overlap DMA with WMMA+softmax).
// grid (S/128, H, B); block 256 = 8 waves; wave owns a 16-row q tile.
// ---------------------------------------------------------------------------
__global__ __launch_bounds__(256)
void flash_attn_kernel(const unsigned short* __restrict__ qh,
                       const unsigned short* __restrict__ kh,
                       const unsigned short* __restrict__ vh,
                       unsigned short* __restrict__ ctx) {
    // element offsets inside one shared block (ushorts)
    constexpr int KT0 = 0;          // 2 x (32 tokens x 64 dims) K tiles
    constexpr int VR0 = 4096;       // 2 x (32 tokens x 64 dims) V row tiles
    constexpr int VT0 = 8192;       // 64 x 32 transposed V tile
    __shared__ unsigned short sm[VT0 + 2048];     // 20 KB

    const int tid  = threadIdx.x;
    const int lane = tid & 31;
    const int wave = tid >> 5;
    const int ln = lane & 15, hs = lane >> 4;
    const int h = blockIdx.y, b = blockIdx.z;
    const int qBase = blockIdx.x * 128 + wave * 16;

    // Q^T B-fragments (lane = N = q-row), loaded once; q is pre-scaled by 1/8
    const unsigned short* qp = qh + ((size_t)(b * Ss) + qBase + ln) * Dm + h * 64;
    Frag16 bQ0, bQ1;
    bQ0.q[0] = *(const uint4*)(qp + hs * 16);
    bQ0.q[1] = *(const uint4*)(qp + hs * 16 + 8);
    bQ1.q[0] = *(const uint4*)(qp + 32 + hs * 16);
    bQ1.q[1] = *(const uint4*)(qp + 32 + hs * 16 + 8);

    v8f acc0 = {}, acc1 = {}, acc2 = {}, acc3 = {};
    float m = -3.0e38f, l = 0.f;

    const unsigned smBase = lds_off(sm);
    const int stgTok = tid >> 3;                  // 0..31
    const int stgEl  = (tid & 7) * 8;             // 0..56

    auto stage = [&](int kb, int bufSel) {
        const unsigned short* kg = kh + ((size_t)(b * Ss) + kb) * Dm + h * 64;
        const unsigned short* vg = vh + ((size_t)(b * Ss) + kb) * Dm + h * 64;
#if HAVE_TDM
        if (wave == 0)                            // one TDM op per workgroup
            tdm_load_2d_bf16(smBase + (unsigned)(KT0 + bufSel * 2048) * 2, kg,
                             64u, 32u, (unsigned long long)Dm, 64u, 32u);
#else
        *(uint4*)&sm[KT0 + bufSel * 2048 + stgTok * 64 + stgEl] =
            *(const uint4*)(kg + (size_t)stgTok * Dm + stgEl);
#endif
        async_load_b128_to_lds(
            smBase + (unsigned)(VR0 + bufSel * 2048 + stgTok * 64 + stgEl) * 2,
            vg + (size_t)stgTok * Dm + stgEl);
        if (kb + 32 < Ss)                         // prefetch tile after next
            __builtin_prefetch(kg + 32 * Dm + (size_t)stgTok * Dm + stgEl, 0, 1);
    };

    stage(0, 0);                                  // prologue: tile 0 in flight

    for (int kb = 0; kb < Ss; kb += 32) {
        const int cur = (kb >> 5) & 1;
        asm volatile("s_wait_asynccnt 0" ::: "memory");
#if HAVE_TDM
        if (wave == 0) __builtin_amdgcn_s_wait_tensorcnt(0);
#endif
        __syncthreads();                          // tile `cur` landed in LDS

        // transpose vRow[cur] -> vT through registers (LDS -> LDS)
        {
            union { uint4 u; unsigned short s[8]; } t;
            t.u = *(const uint4*)&sm[VR0 + cur * 2048 + stgTok * 64 + stgEl];
            #pragma unroll
            for (int j = 0; j < 8; ++j)
                sm[VT0 + (stgEl + j) * 32 + stgTok] = t.s[j];
        }
        __syncthreads();                          // vT visible everywhere

        if (kb + 32 < Ss) stage(kb + 32, cur ^ 1);  // overlap next DMA

        const unsigned short* kTc = sm + KT0 + cur * 2048;

        // ---- S^T tiles: (keys c*16..c*16+15) x (16 q-rows), 4 WMMAs -----
        v8f sc[2];
        #pragma unroll
        for (int c = 0; c < 2; ++c) {
            const int tok = c * 16 + ln;          // A-frag: lane = M = key token
            Frag16 a0, a1;
            a0.q[0] = *(const uint4*)&kTc[tok * 64 + hs * 16];
            a0.q[1] = *(const uint4*)&kTc[tok * 64 + hs * 16 + 8];
            a1.q[0] = *(const uint4*)&kTc[tok * 64 + 32 + hs * 16];
            a1.q[1] = *(const uint4*)&kTc[tok * 64 + 32 + hs * 16 + 8];
            v8f cc = {};
            cc = wmma_bf16(a0, bQ0, cc);
            cc = wmma_bf16(a1, bQ1, cc);
            sc[c] = cc;
        }

        // ---- per-lane online softmax (lane owns q-row `ln`) -------------
        float mx = -3.0e38f;
        #pragma unroll
        for (int r = 0; r < 8; ++r) { mx = fmaxf(mx, sc[0][r]); mx = fmaxf(mx, sc[1][r]); }
        mx = fmaxf(mx, __shfl_xor(mx, 16, 32));   // combine xor-16 partner
        const float mN = fmaxf(m, mx);
        const float alpha = __expf(m - mN);
        m = mN;
        float p0[8], p1[8], sum = 0.f;
        #pragma unroll
        for (int r = 0; r < 8; ++r) {
            p0[r] = __expf(sc[0][r] - mN);
            p1[r] = __expf(sc[1][r] - mN);
            sum += p0[r] + p1[r];
        }
        sum += __shfl_xor(sum, 16, 32);
        l = l * alpha + sum;
        acc0 *= alpha; acc1 *= alpha; acc2 *= alpha; acc3 *= alpha;

        // ---- P^T -> B-fragment: pack bf16 pairs + 4-dword half swap -----
        unsigned k0p[4], k1p[4];
        #pragma unroll
        for (int j = 0; j < 4; ++j) {
            k0p[j] = (unsigned)f2bf(p0[2 * j]) | ((unsigned)f2bf(p0[2 * j + 1]) << 16);
            k1p[j] = (unsigned)f2bf(p1[2 * j]) | ((unsigned)f2bf(p1[2 * j + 1]) << 16);
        }
        Frag16 bP;
        unsigned w[8];
        #pragma unroll
        for (int j = 0; j < 4; ++j) {
            const unsigned send = hs ? k0p[j] : k1p[j];
            const unsigned recv = (unsigned)__shfl_xor((int)send, 16, 32);
            w[j]     = hs ? recv   : k0p[j];
            w[4 + j] = hs ? k1p[j] : recv;
        }
        bP.q[0] = make_uint4(w[0], w[1], w[2], w[3]);
        bP.q[1] = make_uint4(w[4], w[5], w[6], w[7]);

        // ---- ctx^T += V^T (64x32) x P^T (32x16): 4 WMMAs ---------------
        #pragma unroll
        for (int n = 0; n < 4; ++n) {
            const int vd = n * 16 + ln;           // A-frag: lane = M = value dim
            Frag16 aV;
            aV.q[0] = *(const uint4*)&sm[VT0 + vd * 32 + hs * 16];
            aV.q[1] = *(const uint4*)&sm[VT0 + vd * 32 + hs * 16 + 8];
            v8f* ap = (n == 0) ? &acc0 : (n == 1) ? &acc1 : (n == 2) ? &acc2 : &acc3;
            *ap = wmma_bf16(aV, bP, *ap);
        }
    }

    // ---- epilogue: lane owns q-row; 4 coalesced 16B stores --------------
    const float invl = 1.f / l;
    unsigned short* op = ctx + ((size_t)(b * Ss) + qBase + ln) * Dm + h * 64 + hs * 8;
    #pragma unroll
    for (int n = 0; n < 4; ++n) {
        const v8f& acc = (n == 0) ? acc0 : (n == 1) ? acc1 : (n == 2) ? acc2 : acc3;
        unsigned o[4];
        #pragma unroll
        for (int j = 0; j < 4; ++j)
            o[j] = (unsigned)f2bf(acc[2 * j] * invl) |
                   ((unsigned)f2bf(acc[2 * j + 1] * invl) << 16);
        *(uint4*)(op + n * 16) = make_uint4(o[0], o[1], o[2], o[3]);
    }
}

// ---------------------------------------------------------------------------
// Host orchestration
// ---------------------------------------------------------------------------
extern "C" void kernel_launch(void* const* d_in, const int* in_sizes, int n_in,
                              void* d_out, int out_size, void* d_ws, size_t ws_size,
                              hipStream_t stream) {
    (void)in_sizes; (void)n_in; (void)out_size; (void)ws_size;
    const float* q   = (const float*)d_in[0];
    const float* k   = (const float*)d_in[1];
    const float* v   = (const float*)d_in[2];
    const float* Wq  = (const float*)d_in[3];
    const float* bq  = (const float*)d_in[4];
    const float* Wk  = (const float*)d_in[5];
    const float* bk  = (const float*)d_in[6];
    const float* Wv  = (const float*)d_in[7];
    const float* bv  = (const float*)d_in[8];
    const float* Wo  = (const float*)d_in[9];
    const float* bo  = (const float*)d_in[10];

    char* ws = (char*)d_ws;
    auto alloc = [&](size_t bytes) -> void* {
        void* p = ws;
        ws += (bytes + 255) & ~(size_t)255;
        return p;
    };
    const size_t actB = (size_t)TOK * Dm * sizeof(unsigned short);   // 16 MB
    const size_t wB   = (size_t)Dm * Dm * sizeof(unsigned short);    //  2 MB
    unsigned short* qn  = (unsigned short*)alloc(actB);
    unsigned short* kn  = (unsigned short*)alloc(actB);
    unsigned short* vn  = (unsigned short*)alloc(actB);
    unsigned short* WqT = (unsigned short*)alloc(wB);
    unsigned short* WkT = (unsigned short*)alloc(wB);
    unsigned short* WvT = (unsigned short*)alloc(wB);
    unsigned short* WoT = (unsigned short*)alloc(wB);
    unsigned short* qhv = (unsigned short*)alloc(actB);
    unsigned short* khv = (unsigned short*)alloc(actB);
    unsigned short* vhv = (unsigned short*)alloc(actB);
    unsigned short* ctx = (unsigned short*)alloc(actB);

    // 1) weight prep: transpose + bf16
    const int wGrid = (Dm * Dm + 255) / 256;
    transpose_bf16_kernel<<<wGrid, 256, 0, stream>>>(Wq, WqT, Dm, Dm);
    transpose_bf16_kernel<<<wGrid, 256, 0, stream>>>(Wk, WkT, Dm, Dm);
    transpose_bf16_kernel<<<wGrid, 256, 0, stream>>>(Wv, WvT, Dm, Dm);
    transpose_bf16_kernel<<<wGrid, 256, 0, stream>>>(Wo, WoT, Dm, Dm);

    // 2) LN (+RoPE) -> bf16 activations
    ln_rope_kernel<<<TOK, 256, 0, stream>>>(q, qn, 1);
    ln_rope_kernel<<<TOK, 256, 0, stream>>>(k, kn, 1);
    ln_rope_kernel<<<TOK, 256, 0, stream>>>(v, vn, 0);

    // 3) head projections (WMMA); q pre-scaled by 1/sqrt(64)
    dim3 gGrid(TOK / 64, Dm / 64);
    wmma_gemm_kernel<false><<<gGrid, 256, 0, stream>>>(qn, WqT, bq, qhv, TOK, Dm, Dm, 0.125f);
    wmma_gemm_kernel<false><<<gGrid, 256, 0, stream>>>(kn, WkT, bk, khv, TOK, Dm, Dm, 1.0f);
    wmma_gemm_kernel<false><<<gGrid, 256, 0, stream>>>(vn, WvT, bv, vhv, TOK, Dm, Dm, 1.0f);

    // 4) fused flash attention (WMMA + TDM + async-to-LDS, double buffered)
    dim3 fGrid(Ss / 128, Hh, Bb);
    flash_attn_kernel<<<fGrid, 256, 0, stream>>>(qhv, khv, vhv, ctx);

    // 5) output projection (WMMA, f32 out + bias)
    wmma_gemm_kernel<true><<<gGrid, 256, 0, stream>>>(ctx, WoT, bo, d_out, TOK, Dm, Dm, 1.0f);
}
